// SCN1_22488448762087
// MI455X (gfx1250) — matrix operand order repaired and checked by reading, not compiled, not run on hardware
//
#include <hip/hip_runtime.h>

// SCN U-Net forward for MI455X (gfx1250, wave32, WMMA).
// f16 feature staging + fragment-packed f16 weights feeding
// v_wmma_f32_16x16x32_f16; f32 masters for BN stats / residuals; small VALU
// kernels for deconv/concat/head. Gather-GEMM waves process ROWS x 16 rows so
// each B fragment load feeds ROWS WMMAs (B re-fetch traffic dominates at the
// fine levels). All scratch from d_ws bump arena; stream-ordered,
// deterministic, graph-capture safe.

typedef __attribute__((ext_vector_type(16))) _Float16     v16h;
typedef __attribute__((ext_vector_type(8)))  float        v8f;
typedef __attribute__((ext_vector_type(4)))  unsigned int v4u;

#define EPSV 1e-4f

// ----------------------------- kernels -------------------------------------

__global__ __launch_bounds__(256) void k_convert_pad(
    const float* __restrict__ X, int N, int C, int Cp, _Float16* __restrict__ Y)
{
  int i = blockIdx.x * blockDim.x + threadIdx.x;
  int total = N * Cp;
  if (i >= total) return;
  int c = i % Cp, n = i / Cp;
  float v = (c < C) ? X[(size_t)n * C + c] : 0.f;
  Y[i] = (_Float16)v;
}

// Pack W [taps, Cin, Cout] f32 -> fragment-ordered f16:
// Wp[frag*512 + lane*16 + h], frag = (t*KC+kc)*NC + c,
// B layout: col = lane&15, K = (h&7) + 8*(lane>>4) + 16*(h>>3). Pads cin >= Cin with 0.
__global__ __launch_bounds__(256) void k_pack_w(
    const float* __restrict__ W, _Float16* __restrict__ Wp,
    int taps, int Cin, int Cout, int KC, int NC)
{
  size_t total = (size_t)taps * KC * NC * 512;
  for (size_t i = (size_t)blockIdx.x * blockDim.x + threadIdx.x; i < total;
       i += (size_t)gridDim.x * blockDim.x) {
    int h    = (int)(i & 15);
    int lane = (int)((i >> 4) & 31);
    size_t f = i >> 9;
    int c  = (int)(f % NC); f /= NC;
    int kc = (int)(f % KC);
    int t  = (int)(f / KC);
    int K    = (h & 7) + ((lane >> 4) << 3) + ((h >> 3) << 4);
    int cin  = kc * 32 + K;
    int cout = c * 16 + (lane & 15);
    float v = (cin < Cin) ? W[((size_t)t * Cin + cin) * Cout + cout] : 0.f;
    Wp[i] = (_Float16)v;
  }
}

// Deterministic per-channel batch stats: stats[c]=mean, stats[C+c]=rsqrt(var+eps)
__global__ __launch_bounds__(256) void k_bn_stats(
    const float* __restrict__ X, int N, int C, float* __restrict__ stats)
{
  __shared__ float s1[256], s2[256];
  int c = blockIdx.x;
  float s = 0.f, q = 0.f;
  for (int n = threadIdx.x; n < N; n += 256) {
    float v = X[(size_t)n * C + c];
    s += v; q += v * v;
  }
  s1[threadIdx.x] = s; s2[threadIdx.x] = q;
  __syncthreads();
  for (int st = 128; st > 0; st >>= 1) {
    if ((int)threadIdx.x < st) {
      s1[threadIdx.x] += s1[threadIdx.x + st];
      s2[threadIdx.x] += s2[threadIdx.x + st];
    }
    __syncthreads();
  }
  if (threadIdx.x == 0) {
    float mean = s1[0] / (float)N;
    float var  = s2[0] / (float)N - mean * mean;
    stats[c]     = mean;
    stats[C + c] = rsqrtf(var + EPSV);
  }
}

__global__ __launch_bounds__(256) void k_bn_apply(
    const float* __restrict__ X, int N, int C,
    const float* __restrict__ stats,
    const float* __restrict__ gamma, const float* __restrict__ beta,
    _Float16* __restrict__ Y16, float* __restrict__ Y32)
{
  int i = blockIdx.x * blockDim.x + threadIdx.x;
  int total = N * C;
  if (i >= total) return;
  int c = i % C;
  float v = (X[i] - stats[c]) * stats[C + c] * gamma[c] + beta[c];
  v = v > 0.f ? v : 0.f;
  if (Y16) Y16[i] = (_Float16)v;
  if (Y32) Y32[i] = v;
}

// Gather-GEMM: Y[n,d] = sum_t sum_c X[idx[n,t], c] * W[t,c,d]  (+ residual)
// idx==nullptr -> identity (taps must be 1).
// Wave = (ROWS*16) rows x Cout tile; each B fragment feeds ROWS WMMAs.
template <int NC, int ROWS>
__global__ __launch_bounds__(128) void k_gather_conv(
    const _Float16* __restrict__ X, int ldx,
    const int* __restrict__ idx, int taps,
    const _Float16* __restrict__ Wp, int KC,
    const float* __restrict__ residual,
    float* __restrict__ Y, int N, int Cout)
{
  const int lane = threadIdx.x & 31;
  const int wave = threadIdx.x >> 5;
  const int r0   = (blockIdx.x * 4 + wave) * (16 * ROWS);
  if (r0 >= N) return;
  const int mrow = lane & 15;
  const int hi   = lane >> 4;

  v8f zero8 = {0.f, 0.f, 0.f, 0.f, 0.f, 0.f, 0.f, 0.f};
  v8f acc[ROWS][NC];
#pragma unroll
  for (int j = 0; j < ROWS; ++j)
#pragma unroll
    for (int c = 0; c < NC; ++c) acc[j][c] = zero8;

  v4u zero4 = {0u, 0u, 0u, 0u};

  for (int t = 0; t < taps; ++t) {
    int src[ROWS];
#pragma unroll
    for (int j = 0; j < ROWS; ++j) {
      int row = r0 + j * 16 + mrow;
      src[j] = -1;
      if (row < N) src[j] = idx ? idx[(size_t)row * taps + t] : row;
    }
    for (int kc = 0; kc < KC; ++kc) {
      // A frag (16x32 f16): lane<16 holds K=0..7,16..23; lane>=16 K=8..15,24..31
      union { v4u q[2]; v16h v; } a[ROWS];
#pragma unroll
      for (int j = 0; j < ROWS; ++j) {
        if (src[j] >= 0) {
          const v4u* p = (const v4u*)(X + (size_t)src[j] * ldx + kc * 32 + (hi << 3));
          a[j].q[0] = p[0];
          a[j].q[1] = p[2];
        } else {
          a[j].q[0] = zero4;
          a[j].q[1] = zero4;
        }
      }
      const _Float16* wb = Wp + (((size_t)t * KC + kc) * NC) * 512 + lane * 16;
#pragma unroll
      for (int c = 0; c < NC; ++c) {
        union { v4u q[2]; v16h v; } b;
        const v4u* wq = (const v4u*)(wb + (size_t)c * 512);
        b.q[0] = wq[0];
        b.q[1] = wq[1];
#pragma unroll
        for (int j = 0; j < ROWS; ++j) {
          acc[j][c] = __builtin_amdgcn_wmma_f32_16x16x32_f16(
              false, a[j].v, false, b.v, (short)0, acc[j][c], false, false);
        }
      }
    }
  }

  // D frag (16x16 f32): lane holds col = lane&15, rows m = r + 8*(lane>>4)
#pragma unroll
  for (int j = 0; j < ROWS; ++j) {
#pragma unroll
    for (int c = 0; c < NC; ++c) {
#pragma unroll
      for (int r = 0; r < 8; ++r) {
        int m    = r + (hi << 3);
        int orow = r0 + j * 16 + m;
        if (orow < N) {
          size_t o = (size_t)orow * Cout + c * 16 + mrow;
          float v = acc[j][c][r];
          if (residual) v += residual[o];
          Y[o] = v;
        }
      }
    }
  }
}

// Stride-2 deconv scatter side: y[n,d] = sum_c xc[parent[n],c] * W[offset[n],c,d]
__global__ __launch_bounds__(256) void k_deconv(
    const _Float16* __restrict__ Xc, const int* __restrict__ parent,
    const int* __restrict__ offs, const float* __restrict__ Wup,
    float* __restrict__ Y, int Nf, int Cn, int C)
{
  int i = blockIdx.x * blockDim.x + threadIdx.x;
  int total = Nf * C;
  if (i >= total) return;
  int d = i % C, n = i / C;
  int p = parent[n], o = offs[n];
  const _Float16* xr = Xc + (size_t)p * Cn;
  const float*    wr = Wup + (size_t)o * Cn * C + d;
  float acc = 0.f;
  for (int ci = 0; ci < Cn; ++ci)
    acc += (float)xr[ci] * wr[(size_t)ci * C];
  Y[i] = acc;
}

__global__ __launch_bounds__(256) void k_concat(
    const float* __restrict__ Xa, const float* __restrict__ Xb,
    float* __restrict__ O, int N, int C)
{
  int i = blockIdx.x * blockDim.x + threadIdx.x;
  int C2 = 2 * C;
  int total = N * C2;
  if (i >= total) return;
  int c = i % C2, n = i / C2;
  O[i] = (c < C) ? Xa[(size_t)n * C + c] : Xb[(size_t)n * C + (c - C)];
}

__global__ __launch_bounds__(256) void k_head(
    const float* __restrict__ FV, const float* __restrict__ Wl,
    const float* __restrict__ bl, float* __restrict__ Y, int N)
{
  int i = blockIdx.x * blockDim.x + threadIdx.x;
  if (i >= N * 20) return;
  int j = i % 20, n = i / 20;
  float acc = bl[j];
  const float* f = FV + (size_t)n * 32;
#pragma unroll
  for (int c = 0; c < 32; ++c) acc += f[c] * Wl[c * 20 + j];
  Y[i] = acc;
}

// ----------------------------- host side -----------------------------------

namespace {

static const int NP[7] = {32, 64, 96, 128, 160, 192, 224};

struct Arena {
  char* base; size_t off, cap;
  void* alloc(size_t bytes) {
    size_t a = (off + 255) & ~(size_t)255;
    off = a + bytes;
    return (void*)(base + a);
  }
  float*    f32(size_t n) { return (float*)alloc(n * 4); }
  _Float16* f16(size_t n) { return (_Float16*)alloc(n * 2); }
};

struct BNP { const float* g; const float* b; };
struct RBP { BNP bn1, bn2; _Float16 *w1p, *w2p, *ninp; int cin, cout; };
struct LVP {
  RBP enc[2];
  BNP bnd, bnu;
  _Float16* wdp;
  const float* wup;
  RBP dec[2];
};
struct Net {
  int ns[7];
  LVP lv[7];
  const int* neigh[7];
  const int* down[6];
  const int* parent[6];
  const int* offset[6];
};

inline int cdiv(int a, int b) { return (a + b - 1) / b; }

void launch_gather(const _Float16* X, int ldx, const int* idx, int taps,
                   const _Float16* Wp, int KC, const float* res,
                   float* Y, int N, int Cout, hipStream_t s)
{
  dim3 block(128);
  int nc = Cout / 16;
  // ROWS=2 (32 rows/wave) for the register-light fine levels: halves B
  // fragment re-fetch per WMMA. ROWS=1 for wide-accumulator coarse levels.
  if (nc <= 8) {
    dim3 grid(cdiv(N, 128));
    switch (nc) {
      case 2: k_gather_conv<2, 2><<<grid, block, 0, s>>>(X, ldx, idx, taps, Wp, KC, res, Y, N, Cout); break;
      case 4: k_gather_conv<4, 2><<<grid, block, 0, s>>>(X, ldx, idx, taps, Wp, KC, res, Y, N, Cout); break;
      case 6: k_gather_conv<6, 2><<<grid, block, 0, s>>>(X, ldx, idx, taps, Wp, KC, res, Y, N, Cout); break;
      case 8: k_gather_conv<8, 2><<<grid, block, 0, s>>>(X, ldx, idx, taps, Wp, KC, res, Y, N, Cout); break;
      default: break;
    }
  } else {
    dim3 grid(cdiv(N, 64));
    switch (nc) {
      case 10: k_gather_conv<10, 1><<<grid, block, 0, s>>>(X, ldx, idx, taps, Wp, KC, res, Y, N, Cout); break;
      case 12: k_gather_conv<12, 1><<<grid, block, 0, s>>>(X, ldx, idx, taps, Wp, KC, res, Y, N, Cout); break;
      case 14: k_gather_conv<14, 1><<<grid, block, 0, s>>>(X, ldx, idx, taps, Wp, KC, res, Y, N, Cout); break;
      default: break;
    }
  }
}

_Float16* pack_w(Arena& A, const float* W, int taps, int Cin, int Cout,
                 hipStream_t s)
{
  int KC = cdiv(Cin, 32), NC = Cout / 16;
  size_t total = (size_t)taps * KC * NC * 512;
  _Float16* Wp = A.f16(total);
  int blocks = (int)((total + 255) / 256);
  if (blocks > 2048) blocks = 2048;
  k_pack_w<<<blocks, 256, 0, s>>>(W, Wp, taps, Cin, Cout, KC, NC);
  return Wp;
}

// pre-activation residual block: out = shortcut(x) + conv2(bnrelu(conv1(bnrelu(x))))
void res_block(Arena& A, const float* x, int N, int Cin, int Cout,
               const RBP& p, const int* neigh, float* out, hipStream_t s)
{
  size_t m0 = A.off;
  float* h1 = A.f32((size_t)N * Cout);
  float* sbuf = nullptr;
  if (p.ninp) sbuf = A.f32((size_t)N * Cout);
  {
    size_t m1 = A.off;
    float* st = A.f32(2 * (size_t)Cin);
    _Float16* t16 = A.f16((size_t)N * Cin);
    k_bn_stats<<<Cin, 256, 0, s>>>(x, N, Cin, st);
    k_bn_apply<<<cdiv(N * Cin, 256), 256, 0, s>>>(x, N, Cin, st, p.bn1.g, p.bn1.b, t16, (float*)nullptr);
    launch_gather(t16, Cin, neigh, 27, p.w1p, Cin / 32, nullptr, h1, N, Cout, s);
    A.off = m1;
  }
  const float* shortcut = x;
  if (p.ninp) {
    size_t m1 = A.off;
    _Float16* x16 = A.f16((size_t)N * Cin);
    k_convert_pad<<<cdiv(N * Cin, 256), 256, 0, s>>>(x, N, Cin, Cin, x16);
    launch_gather(x16, Cin, nullptr, 1, p.ninp, Cin / 32, nullptr, sbuf, N, Cout, s);
    shortcut = sbuf;
    A.off = m1;
  }
  {
    size_t m1 = A.off;
    float* st = A.f32(2 * (size_t)Cout);
    _Float16* h16 = A.f16((size_t)N * Cout);
    k_bn_stats<<<Cout, 256, 0, s>>>(h1, N, Cout, st);
    k_bn_apply<<<cdiv(N * Cout, 256), 256, 0, s>>>(h1, N, Cout, st, p.bn2.g, p.bn2.b, h16, (float*)nullptr);
    launch_gather(h16, Cout, neigh, 27, p.w2p, Cout / 32, shortcut, out, N, Cout, s);
    A.off = m1;
  }
  A.off = m0;  // h1/sbuf dead: all consumers already enqueued (in-order stream)
}

float* unet_rec(Arena& A, int l, float* x, const Net& net, hipStream_t s)
{
  int N = net.ns[l], C = NP[l];
  float* cur = x;
  for (int r = 0; r < 2; ++r) {
    float* nx = A.f32((size_t)N * C);
    res_block(A, cur, N, C, C, net.lv[l].enc[r], net.neigh[l], nx, s);
    cur = nx;
  }
  if (l < 6) {
    int Nc = net.ns[l + 1], Cn = NP[l + 1];
    float* yup = A.f32((size_t)N * C);
    size_t mC = A.off;
    float* yc = A.f32((size_t)Nc * Cn);
    {
      size_t m1 = A.off;
      float* st = A.f32(2 * (size_t)C);
      _Float16* t16 = A.f16((size_t)N * C);
      k_bn_stats<<<C, 256, 0, s>>>(cur, N, C, st);
      k_bn_apply<<<cdiv(N * C, 256), 256, 0, s>>>(cur, N, C, st, net.lv[l].bnd.g, net.lv[l].bnd.b, t16, (float*)nullptr);
      launch_gather(t16, C, net.down[l], 8, net.lv[l].wdp, C / 32, nullptr, yc, Nc, Cn, s);
      A.off = m1;
    }
    float* yo = unet_rec(A, l + 1, yc, net, s);
    {
      float* st = A.f32(2 * (size_t)Cn);
      _Float16* u16 = A.f16((size_t)Nc * Cn);
      k_bn_stats<<<Cn, 256, 0, s>>>(yo, Nc, Cn, st);
      k_bn_apply<<<cdiv(Nc * Cn, 256), 256, 0, s>>>(yo, Nc, Cn, st, net.lv[l].bnu.g, net.lv[l].bnu.b, u16, (float*)nullptr);
      k_deconv<<<cdiv(N * C, 256), 256, 0, s>>>(u16, net.parent[l], net.offset[l], net.lv[l].wup, yup, N, Cn, C);
    }
    A.off = mC;  // free entire child frame + down/up transients
    float* cat = A.f32((size_t)N * 2 * C);
    k_concat<<<cdiv(N * 2 * C, 256), 256, 0, s>>>(cur, yup, cat, N, C);
    float* d0 = A.f32((size_t)N * C);
    res_block(A, cat, N, 2 * C, C, net.lv[l].dec[0], net.neigh[l], d0, s);
    float* d1 = A.f32((size_t)N * C);
    res_block(A, d0, N, C, C, net.lv[l].dec[1], net.neigh[l], d1, s);
    cur = d1;
  }
  return cur;
}

}  // namespace

extern "C" void kernel_launch(void* const* d_in, const int* in_sizes, int n_in,
                              void* d_out, int out_size, void* d_ws, size_t ws_size,
                              hipStream_t stream)
{
  (void)in_sizes; (void)n_in; (void)out_size;
  Arena A{(char*)d_ws, 0, ws_size};
  Net net;
  net.ns[0] = 80000;
  for (int i = 1; i < 7; ++i) {
    int v = net.ns[i - 1] / 4;
    net.ns[i] = v < 16 ? 16 : v;
  }

  // Inputs in setup_inputs() insertion order.
  int cur = 0;
  auto F = [&]() { return (const float*)d_in[cur++]; };
  auto I = [&]() { return (const int*)d_in[cur++]; };

  const float* x_raw = F();  // [80000,3]
  const float* w_in  = F();  // [27,3,32]

  for (int l = 0; l < 7; ++l) {
    int c = NP[l];
    for (int r = 0; r < 2; ++r) {
      RBP& p = net.lv[l].enc[r];
      p.bn1.g = F(); p.bn1.b = F();
      const float* w1 = F();
      p.bn2.g = F(); p.bn2.b = F();
      const float* w2 = F();
      p.cin = c; p.cout = c;
      p.w1p = pack_w(A, w1, 27, c, c, stream);
      p.w2p = pack_w(A, w2, 27, c, c, stream);
      p.ninp = nullptr;
    }
    if (l < 6) {
      int cn = NP[l + 1];
      net.lv[l].bnd.g = F(); net.lv[l].bnd.b = F();
      const float* wd = F();
      net.lv[l].bnu.g = F(); net.lv[l].bnu.b = F();
      net.lv[l].wup = F();
      net.lv[l].wdp = pack_w(A, wd, 8, c, cn, stream);
      for (int r = 0; r < 2; ++r) {
        RBP& p = net.lv[l].dec[r];
        int cin = (r == 0) ? 2 * c : c;
        p.bn1.g = F(); p.bn1.b = F();
        const float* w1 = F();
        p.bn2.g = F(); p.bn2.b = F();
        const float* w2 = F();
        p.cin = cin; p.cout = c;
        p.w1p = pack_w(A, w1, 27, cin, c, stream);
        p.w2p = pack_w(A, w2, 27, c, c, stream);
        p.ninp = nullptr;
        if (r == 0) {
          const float* nin = F();
          p.ninp = pack_w(A, nin, 1, cin, c, stream);
        }
      }
    }
  }
  BNP bn_out;
  bn_out.g = F(); bn_out.b = F();
  const float* w_lin = F();
  const float* b_lin = F();
  for (int l = 0; l < 7; ++l) net.neigh[l]  = I();
  for (int l = 0; l < 6; ++l) net.down[l]   = I();
  for (int l = 0; l < 6; ++l) net.parent[l] = I();
  for (int l = 0; l < 6; ++l) net.offset[l] = I();

  // Input submanifold conv 3 -> 32 (Cin padded to 32 with zeros in x16 + Wp).
  _Float16* wip = pack_w(A, w_in, 27, 3, 32, stream);
  int N0 = net.ns[0];
  _Float16* x16 = A.f16((size_t)N0 * 32);
  k_convert_pad<<<cdiv(N0 * 32, 256), 256, 0, stream>>>(x_raw, N0, 3, 32, x16);
  float* h0 = A.f32((size_t)N0 * 32);
  launch_gather(x16, 32, net.neigh[0], 27, wip, 1, nullptr, h0, N0, 32, stream);

  float* h = unet_rec(A, 0, h0, net, stream);

  // fv = bn_relu(h, bn_out) -> d_out[0 : 80000*32] (f32 master path)
  // y  = fv @ w_lin + b     -> d_out[80000*32 : +80000*20]
  float* fv = (float*)d_out;
  float* yo = fv + (size_t)N0 * 32;
  {
    float* st = A.f32(2 * 32);
    k_bn_stats<<<32, 256, 0, stream>>>(h, N0, 32, st);
    k_bn_apply<<<cdiv(N0 * 32, 256), 256, 0, stream>>>(
        h, N0, 32, st, bn_out.g, bn_out.b, (_Float16*)nullptr, fv);
    k_head<<<cdiv(N0 * 20, 256), 256, 0, stream>>>(fv, w_lin, b_lin, yo, N0);
  }
}
